// GausskeyL2P_17815524344306
// MI455X (gfx1250) — compile-verified, matrix operand order checked by reading, not run on hardware
//
#include <hip/hip_runtime.h>
#include <hip/hip_bf16.h>

// ---------------------------------------------------------------------------
// Types for CDNA5 WMMA (gfx1250, wave32)
// ---------------------------------------------------------------------------
typedef __bf16 bf16;
typedef __attribute__((ext_vector_type(16))) __bf16 v16bf;
typedef __attribute__((ext_vector_type(8)))  float  v8f;

union AFrag { v16bf v; bf16 e[16]; };
union CFrag { v8f  v; float e[8]; };
union A16u  { uint4 u4[2]; bf16 h[16]; };
union B8u   { uint4 u4; unsigned u[4]; bf16 h[8]; };
union U2u   { uint2 u; bf16 h[4]; };
union PUu   { unsigned u; bf16 h[2]; };

__device__ inline bf16 to_bf(float v) { return (bf16)v; }
__device__ inline bf16 to_bf(bf16 v) { return v; }

__device__ inline float gelu_tanh(float x) {
  float x3 = x * x * x;
  return 0.5f * x * (1.0f + tanhf(0.7978845608028654f * (x + 0.044715f * x3)));
}

__device__ inline float blockReduceSum256(float v, float* red) {
  int tid = threadIdx.x;
  red[tid] = v; __syncthreads();
  for (int s = 128; s > 0; s >>= 1) { if (tid < s) red[tid] += red[tid + s]; __syncthreads(); }
  float r = red[0]; __syncthreads();
  return r;
}

// ---------------------------------------------------------------------------
// Constants
// ---------------------------------------------------------------------------
#define BB   16
#define DD   768
#define NLAY 12
#define NHEAD 12
#define DHEAD 64
#define DFFN 3072
#define NPAD 224      // padded token count (197 and 222 both <= 224 = 14*16)
#define NPOOL 10
#define STOP 5
#define LPR  5
#define NCLS 100
#define LOG2PI_F 1.8378770664093453f

// ---------------------------------------------------------------------------
// Workhorse: bf16 WMMA GEMM, C[M,N] = act(A[M,K] * B[K,N] + bias) (+ residual)
// Block tile 128x64x32, 8 waves (wave32), each wave 32x32 via 2x2 WMMA frags.
// Double-buffered LDS with register staging: one barrier per K-step; global
// loads for tile kt+1 issue before the WMMAs of tile kt and complete after.
// REQUIRES: K % 32 == 0 (true for every call site: 64/224/768/3072).
// AT: float (cvt to bf16 at LDS store) or bf16.  TRANSB: B stored [N,K].
// ---------------------------------------------------------------------------
template<typename AT, bool TRANSB, int ACT, bool RESID>
__global__ __launch_bounds__(256) void k_gemm(
    const AT* __restrict__ A, const bf16* __restrict__ Bm,
    const float* __restrict__ bias, const float* __restrict__ Rin,
    float* __restrict__ C,
    int M, int N, int K, int lda, int ldb, int ldc,
    long sA, long sB, long sC)
{
  constexpr int BM = 128, BN = 64, BK = 32;
  constexpr bool AF = (sizeof(AT) == 4);
  __shared__ bf16 As[2][BM][BK];   // [m][k]
  __shared__ bf16 Bs[2][BN][BK];   // [n][k]

  const int z = blockIdx.z;
  A  += (long)z * sA;
  Bm += (long)z * sB;
  C  += (long)z * sC;
  const float* R = RESID ? (Rin + (long)z * sC) : nullptr;

  const int bm = blockIdx.y * BM, bn = blockIdx.x * BN;
  const int tid = threadIdx.x, lane = tid & 31, wave = tid >> 5;
  const int wm = (wave >> 1) * 32, wn = (wave & 1) * 32;   // 4x2 wave grid
  const int half = lane >> 4, r = lane & 15;

  // Per-thread staging slots (contiguous, vector-width friendly):
  //  A: 16 elems = half a row:  row arow = tid>>1, cols acol..acol+15
  //  B (normal): 2 K-rows x 4 N-cols;  B (TRANSB): 1 N-row x 8 K-cols
  const int arow = tid >> 1, acol = (tid & 1) * 16;
  const int gmA = bm + arow;
  int bn0, bk0;
  if constexpr (TRANSB) { bn0 = tid >> 2; bk0 = (tid & 3) * 8; }
  else                  { bk0 = (tid >> 4) * 2; bn0 = (tid & 15) * 4; }

  float4 afr[4];          // AF: raw f32 tile regs (cvt deferred to LDS store)
  uint4  abr[2];          // !AF: raw bf16 tile regs
  uint4  btrr;            // TRANSB B regs
  uint2  bnr0, bnr1;      // normal  B regs

  auto load_tile = [&](int k0) {
    if constexpr (AF) {
      if (gmA < M) {
        const float* s = A + (long)gmA * lda + k0 + acol;
        afr[0] = *(const float4*)(s);
        afr[1] = *(const float4*)(s + 4);
        afr[2] = *(const float4*)(s + 8);
        afr[3] = *(const float4*)(s + 12);
      } else {
        afr[0] = afr[1] = afr[2] = afr[3] = float4{0.f, 0.f, 0.f, 0.f};
      }
    } else {
      if (gmA < M) {
        const bf16* s = (const bf16*)A + (long)gmA * lda + k0 + acol;
        abr[0] = *(const uint4*)(s);
        abr[1] = *(const uint4*)(s + 8);
      } else {
        abr[0] = abr[1] = uint4{0u, 0u, 0u, 0u};
      }
    }
    if constexpr (TRANSB) {
      int gn = bn + bn0;
      if (gn < N) btrr = *(const uint4*)(Bm + (long)gn * ldb + k0 + bk0);
      else        btrr = uint4{0u, 0u, 0u, 0u};
    } else {
      int gn = bn + bn0;
      if (gn + 3 < N) {
        const bf16* s = Bm + (long)(k0 + bk0) * ldb + gn;
        bnr0 = *(const uint2*)(s);
        bnr1 = *(const uint2*)(s + ldb);
      } else {              // ragged N edge (head GEMM): elementwise guarded
        U2u r0, r1;
#pragma unroll
        for (int j = 0; j < 4; ++j) {
          int g = gn + j;
          r0.h[j] = (g < N) ? Bm[(long)(k0 + bk0) * ldb + g] : (bf16)0.0f;
          r1.h[j] = (g < N) ? Bm[(long)(k0 + bk0 + 1) * ldb + g] : (bf16)0.0f;
        }
        bnr0 = r0.u; bnr1 = r1.u;
      }
    }
  };

  auto store_tile = [&](int buf) {
    bf16* da = &As[buf][arow][acol];
    if constexpr (AF) {
      A16u t;
#pragma unroll
      for (int i = 0; i < 4; ++i) {
        t.h[4 * i + 0] = (bf16)afr[i].x; t.h[4 * i + 1] = (bf16)afr[i].y;
        t.h[4 * i + 2] = (bf16)afr[i].z; t.h[4 * i + 3] = (bf16)afr[i].w;
      }
      *(uint4*)(da) = t.u4[0];
      *(uint4*)(da + 8) = t.u4[1];
    } else {
      *(uint4*)(da) = abr[0];
      *(uint4*)(da + 8) = abr[1];
    }
    if constexpr (TRANSB) {
      *(uint4*)&Bs[buf][bn0][bk0] = btrr;
    } else {
      U2u r0, r1; r0.u = bnr0; r1.u = bnr1;
#pragma unroll
      for (int j = 0; j < 4; ++j) {
        PUu p; p.h[0] = r0.h[j]; p.h[1] = r1.h[j];   // pair (k, k+1) for col n
        *(unsigned*)&Bs[buf][bn0 + j][bk0] = p.u;
      }
    }
  };

  CFrag acc[2][2];
#pragma unroll
  for (int i = 0; i < 2; ++i)
#pragma unroll
    for (int j = 0; j < 2; ++j)
#pragma unroll
      for (int v = 0; v < 8; ++v) acc[i][j].e[v] = 0.0f;

  const int nk = K / BK;   // K % 32 == 0 by construction
  load_tile(0);
  store_tile(0);
  int cur = 0;
  for (int kt = 0; kt < nk; ++kt) {
    __syncthreads();
    const bool haveNext = (kt + 1 < nk);
    if (haveNext) load_tile((kt + 1) * BK);     // overlaps with WMMAs below
    if (kt + 2 < nk && gmA < M) {               // warm L2 (global_prefetch_b8)
      if constexpr (AF)
        __builtin_prefetch(A + (long)gmA * lda + (kt + 2) * BK + acol, 0, 1);
      else
        __builtin_prefetch((const bf16*)A + (long)gmA * lda + (kt + 2) * BK + acol, 0, 1);
    }

    // Fragment assembly per ISA 7.12.2 16-bit layout: lane r (0-15) is row/col,
    // lane half selects K sets {0..7,16..23}/{8..15,24..31}; two b128 LDS loads each.
    AFrag af[2], bfr[2];
#pragma unroll
    for (int t = 0; t < 2; ++t) {
      int mrow = wm + t * 16 + r;
      int ncol = wn + t * 16 + r;
#pragma unroll
      for (int v = 0; v < 8; ++v) {
        int kb = (v < 4) ? (half * 8 + 2 * v) : (16 + half * 8 + 2 * (v - 4));
        af[t].e[2 * v]      = As[cur][mrow][kb];
        af[t].e[2 * v + 1]  = As[cur][mrow][kb + 1];
        bfr[t].e[2 * v]     = Bs[cur][ncol][kb];
        bfr[t].e[2 * v + 1] = Bs[cur][ncol][kb + 1];
      }
    }
#pragma unroll
    for (int i = 0; i < 2; ++i)
#pragma unroll
      for (int j = 0; j < 2; ++j)
        acc[i][j].v = __builtin_amdgcn_wmma_f32_16x16x32_bf16(
            false, af[i].v, false, bfr[j].v, (short)0, acc[i][j].v, false, false);

    if (haveNext) store_tile(cur ^ 1);          // waits for loads *after* WMMAs
    cur ^= 1;
  }

  // Epilogue: bias / GELU / residual, f32 out. C/D layout: VGPR v -> M = half*8+v, N = r.
#pragma unroll
  for (int i = 0; i < 2; ++i)
#pragma unroll
    for (int j = 0; j < 2; ++j)
#pragma unroll
      for (int v = 0; v < 8; ++v) {
        int gm = bm + wm + i * 16 + half * 8 + v;
        int gn = bn + wn + j * 16 + r;
        if (gm < M && gn < N) {
          float xv = acc[i][j].e[v];
          if (bias) xv += bias[gn];
          if constexpr (ACT == 1) xv = gelu_tanh(xv);
          if constexpr (RESID)    xv += R[(long)gm * ldc + gn];
          C[(long)gm * ldc + gn] = xv;
        }
      }
}

// ---------------------------------------------------------------------------
// Elementwise / reduction kernels
// ---------------------------------------------------------------------------
__global__ void k_cvt(const float* __restrict__ src, bf16* __restrict__ dst, long n) {
  long i = (long)blockIdx.x * blockDim.x + threadIdx.x;
  if (i < n) dst[i] = (bf16)src[i];
}

// src [R,C] row-major -> dst [C,R] bf16 (for patch_w: [D, cyx] -> [cyx, D])
__global__ void k_cvt_T(const float* __restrict__ src, bf16* __restrict__ dst, int Rr, int Cc) {
  long i = (long)blockIdx.x * blockDim.x + threadIdx.x;
  long tot = (long)Rr * Cc;
  if (i >= tot) return;
  int c = (int)(i % Cc); int rr = (int)(i / Cc);
  dst[(long)c * Rr + rr] = (bf16)src[i];
}

__global__ __launch_bounds__(256) void k_layernorm(
    const float* __restrict__ x, const float* __restrict__ g,
    const float* __restrict__ b, float* __restrict__ y, int Dn)
{
  __shared__ float red[256];
  long row = blockIdx.x;
  const float* xr = x + row * Dn;
  float* yr = y + row * Dn;
  int tid = threadIdx.x;
  float s = 0.f;
  for (int i = tid; i < Dn; i += 256) s += xr[i];
  float mean = blockReduceSum256(s, red) / (float)Dn;
  float v = 0.f;
  for (int i = tid; i < Dn; i += 256) { float d = xr[i] - mean; v += d * d; }
  float var = blockReduceSum256(v, red) / (float)Dn;
  float inv = rsqrtf(var + 1e-6f);
  for (int i = tid; i < Dn; i += 256) yr[i] = (xr[i] - mean) * inv * g[i] + b[i];
}

// qkv [B,NPAD,3*D] -> Q,K,V bf16 [B*H, NPAD, DH]; Q pre-scaled; pad rows zeroed
__global__ void k_qkv_split(const float* __restrict__ qkv,
                            bf16* __restrict__ Q, bf16* __restrict__ Kb, bf16* __restrict__ V,
                            int Nvalid, float scale)
{
  long idx = (long)blockIdx.x * blockDim.x + threadIdx.x;
  long total = (long)BB * NHEAD * NPAD * DHEAD;
  if (idx >= total) return;
  int d = (int)(idx & 63); long t = idx >> 6;
  int n = (int)(t % NPAD); t /= NPAD;
  int h = (int)(t % NHEAD); int b = (int)(t / NHEAD);
  float qv = 0.f, kv = 0.f, vv = 0.f;
  if (n < Nvalid) {
    long base = ((long)b * NPAD + n) * (3 * DD) + h * DHEAD + d;
    qv = qkv[base]; kv = qkv[base + DD]; vv = qkv[base + 2 * DD];
  }
  long o = (((long)b * NHEAD + h) * NPAD + n) * DHEAD + d;
  Q[o] = (bf16)(qv * scale); Kb[o] = (bf16)kv; V[o] = (bf16)vv;
}

__global__ __launch_bounds__(256) void k_softmax_bf16(
    const float* __restrict__ S, bf16* __restrict__ P, int Nvalid)
{
  __shared__ float red[256];
  long row = blockIdx.x;
  const float* sr = S + row * NPAD;
  bf16* pr = P + row * NPAD;
  int tid = threadIdx.x;
  float m = -3.4e38f;
  for (int j = tid; j < Nvalid; j += 256) m = fmaxf(m, sr[j]);
  red[tid] = m; __syncthreads();
  for (int s2 = 128; s2 > 0; s2 >>= 1) { if (tid < s2) red[tid] = fmaxf(red[tid], red[tid + s2]); __syncthreads(); }
  m = red[0]; __syncthreads();
  float sum = 0.f;
  for (int j = tid; j < Nvalid; j += 256) sum += __expf(sr[j] - m);
  sum = blockReduceSum256(sum, red);
  float inv = 1.0f / sum;
  for (int j = tid; j < NPAD; j += 256)
    pr[j] = (j < Nvalid) ? (bf16)(__expf(sr[j] - m) * inv) : (bf16)0.0f;
}

// O [B*H, NPAD, DH] -> ctx [B, NPAD, D]; pad rows zeroed
__global__ void k_merge_heads(const float* __restrict__ O, float* __restrict__ ctx, int Nvalid) {
  long idx = (long)blockIdx.x * blockDim.x + threadIdx.x;
  long total = (long)BB * NPAD * DD;
  if (idx >= total) return;
  int c = (int)(idx % DD); long t = idx / DD;
  int n = (int)(t % NPAD); int b = (int)(t / NPAD);
  int h = c >> 6, d = c & 63;
  ctx[idx] = (n < Nvalid) ? O[(((long)b * NHEAD + h) * NPAD + n) * DHEAD + d] : 0.0f;
}

// im2col for 16x16/stride16 patch conv: inputs [B,3,224,224] -> Ap [B*196, 768]
__global__ void k_im2col(const float* __restrict__ img, float* __restrict__ Ap) {
  long idx = (long)blockIdx.x * blockDim.x + threadIdx.x;
  long total = (long)BB * 196 * DD;
  if (idx >= total) return;
  int col = (int)(idx % DD); long row = idx / DD;
  int p = (int)(row % 196); int b = (int)(row / 196);
  int c = col >> 8, yy = (col >> 4) & 15, xx = col & 15;
  int py = p / 14, px = p % 14;
  Ap[idx] = img[(((long)b * 3 + c) * 224 + (py * 16 + yy)) * 224 + (px * 16 + xx)];
}

__global__ void k_build_x0(const float* __restrict__ imgTok, const float* __restrict__ cls,
                           const float* __restrict__ pos, float* __restrict__ x) {
  long idx = (long)blockIdx.x * blockDim.x + threadIdx.x;
  long total = (long)BB * NPAD * DD;
  if (idx >= total) return;
  int c = (int)(idx % DD); long t = idx / DD;
  int n = (int)(t % NPAD); int b = (int)(t / NPAD);
  float v;
  if (n == 0)        v = cls[c] + pos[c];
  else if (n <= 196) v = imgTok[((long)b * 196 + (n - 1)) * DD + c] + pos[(long)n * DD + c];
  else               v = 0.0f;
  x[idx] = v;
}

__global__ void k_build_x2(const float* __restrict__ imgTok, const float* __restrict__ cls,
                           const float* __restrict__ pos, const float* __restrict__ prompts,
                           float* __restrict__ x) {
  long idx = (long)blockIdx.x * blockDim.x + threadIdx.x;
  long total = (long)BB * NPAD * DD;
  if (idx >= total) return;
  int c = (int)(idx % DD); long t = idx / DD;
  int n = (int)(t % NPAD); int b = (int)(t / NPAD);
  float v;
  if (n == 0)             v = cls[c] + pos[c];
  else if (n <= STOP*LPR) v = prompts[((long)b * (STOP * LPR) + (n - 1)) * DD + c];
  else if (n <= STOP*LPR + 196) v = imgTok[((long)b * 196 + (n - 1 - STOP * LPR)) * DD + c];
  else                    v = 0.0f;
  x[idx] = v;
}

// Column-by-column Cholesky of cov = |var| + I, one workgroup per pool component.
__global__ __launch_bounds__(256) void k_cholesky(
    const float* __restrict__ var, float* __restrict__ L, float* __restrict__ logdet)
{
  __shared__ float rowj[DD];
  __shared__ float red[256];
  __shared__ float s_ljj;
  int p = blockIdx.x, tid = threadIdx.x;
  const float* Ain = var + (long)p * DD * DD;
  float* Lp = L + (long)p * DD * DD;
  for (long i = tid; i < (long)DD * DD; i += 256) {
    int r = (int)(i / DD), c = (int)(i % DD);
    float a = fabsf(Ain[i]) + (r == c ? 1.0f : 0.0f);
    Lp[i] = (c <= r) ? a : 0.0f;
  }
  __syncthreads();
  float ld = 0.0f;
  for (int j = 0; j < DD; ++j) {
    float s = 0.f;
    for (int k = tid; k < j; k += 256) { float v = Lp[(long)j * DD + k]; s += v * v; }
    s = blockReduceSum256(s, red);
    if (tid == 0) {
      float ljj = sqrtf(fmaxf(Lp[(long)j * DD + j] - s, 1e-20f));
      Lp[(long)j * DD + j] = ljj; s_ljj = ljj; ld += 2.0f * logf(ljj);
    }
    __syncthreads();
    for (int k = tid; k < j; k += 256) rowj[k] = Lp[(long)j * DD + k];
    __syncthreads();
    float inv = 1.0f / s_ljj;
    for (int i = j + 1 + tid; i < DD; i += 256) {
      float dot = 0.f;
      for (int k = 0; k < j; ++k) dot += Lp[(long)i * DD + k] * rowj[k];
      Lp[(long)i * DD + j] = (Lp[(long)i * DD + j] - dot) * inv;
    }
    __syncthreads();
  }
  if (tid == 0) logdet[p] = ld;
}

// MVN log-prob via forward substitution; one workgroup per (b, p).
__global__ __launch_bounds__(256) void k_mvn(
    const float* __restrict__ hq, long qstride, const float* __restrict__ meanp,
    const float* __restrict__ L, const float* __restrict__ logdet, float* __restrict__ logp)
{
  __shared__ float y[DD];
  __shared__ float diff[DD];
  __shared__ float red[256];
  __shared__ float s_quad;
  int b = blockIdx.x / NPOOL, p = blockIdx.x % NPOOL;
  int tid = threadIdx.x;
  const float* q = hq + (long)b * qstride;
  const float* mu = meanp + (long)p * DD;
  const float* Lp = L + (long)p * DD * DD;
  for (int i = tid; i < DD; i += 256) diff[i] = q[i] - mu[i];
  if (tid == 0) s_quad = 0.f;
  __syncthreads();
  for (int i = 0; i < DD; ++i) {
    float s = 0.f;
    for (int k = tid; k < i; k += 256) s += Lp[(long)i * DD + k] * y[k];
    s = blockReduceSum256(s, red);
    if (tid == 0) {
      float yi = (diff[i] - s) / Lp[(long)i * DD + i];
      y[i] = yi; s_quad += yi * yi;
    }
    __syncthreads();
  }
  if (tid == 0)
    logp[b * NPOOL + p] = -0.5f * ((float)DD * LOG2PI_F + logdet[p] + s_quad);
}

// Top-5 of 10 + gather prompt tokens (+ pos_embed[0,0]); one workgroup per batch row.
__global__ __launch_bounds__(256) void k_topk_gather(
    const float* __restrict__ logp, const float* __restrict__ prompt,
    const float* __restrict__ pos00, float* __restrict__ prompts)
{
  __shared__ int sel[STOP];
  int b = blockIdx.x, tid = threadIdx.x;
  if (tid == 0) {
    float lp[NPOOL];
    for (int i = 0; i < NPOOL; ++i) lp[i] = logp[b * NPOOL + i];
    for (int s = 0; s < STOP; ++s) {
      int am = 0; float bv = -3.4e38f;
      for (int i = 0; i < NPOOL; ++i) if (lp[i] > bv) { bv = lp[i]; am = i; }
      sel[s] = am; lp[am] = -3.4e38f;
    }
  }
  __syncthreads();
  for (int idx = tid; idx < STOP * LPR * DD; idx += 256) {
    int c = idx % DD; int t = idx / DD; int s = t / LPR; int l = t % LPR;
    prompts[((long)b * (STOP * LPR) + t) * DD + c] =
        prompt[((long)sel[s] * LPR + l) * DD + c] + pos00[c];
  }
}

// Mean over prompt tokens 1..S*LP of final LN output -> pooled [B, D]
__global__ void k_pool(const float* __restrict__ h, float* __restrict__ pooled) {
  long idx = (long)blockIdx.x * blockDim.x + threadIdx.x;
  if (idx >= (long)BB * DD) return;
  int c = (int)(idx % DD); int b = (int)(idx / DD);
  float s = 0.f;
  for (int t = 1; t <= STOP * LPR; ++t) s += h[((long)b * NPAD + t) * DD + c];
  pooled[idx] = s * (1.0f / (STOP * LPR));
}

// ---------------------------------------------------------------------------
// Host-side helpers
// ---------------------------------------------------------------------------
static inline char* bump(char*& p, size_t bytes) {
  char* r = p;
  p += (bytes + 255) & ~(size_t)255;
  return r;
}

template<typename AT, bool TRANSB, int ACT, bool RESID>
static void launch_gemm(const void* A, const bf16* Bm, const float* bias, const float* R,
                        float* C, int M, int N, int K, int lda, int ldb, int ldc,
                        int batch, long sA, long sB, long sC, hipStream_t st) {
  dim3 grid((unsigned)((N + 63) / 64), (unsigned)((M + 127) / 128), (unsigned)batch);
  k_gemm<AT, TRANSB, ACT, RESID><<<grid, 256, 0, st>>>(
      (const AT*)A, Bm, bias, R, C, M, N, K, lda, ldb, ldc, sA, sB, sC);
}

struct PassCtx {
  const float *ln1_g, *ln1_b, *qkv_b, *proj_b, *ln2_g, *ln2_b, *fc1_b, *fc2_b;
  const bf16 *wqkv, *wproj, *wfc1, *wfc2;
  float *x, *h, *qkvbuf, *ctxbuf, *mid, *obuf, *scores;
  bf16 *Qb, *Kb, *Vb, *attb;
  hipStream_t st;
};

static void run_pass(const PassCtx& c, int Nvalid) {
  const int M = BB * NPAD;
  const long qkvTot = (long)BB * NHEAD * NPAD * DHEAD;
  const long mhTot  = (long)BB * NPAD * DD;
  for (int l = 0; l < NLAY; ++l) {
    k_layernorm<<<M, 256, 0, c.st>>>(c.x, c.ln1_g + l * DD, c.ln1_b + l * DD, c.h, DD);
    launch_gemm<float, false, 0, false>(c.h, c.wqkv + (long)l * DD * 3 * DD,
        c.qkv_b + l * 3 * DD, nullptr, c.qkvbuf, M, 3 * DD, DD, DD, 3 * DD, 3 * DD,
        1, 0, 0, 0, c.st);
    k_qkv_split<<<(unsigned)((qkvTot + 255) / 256), 256, 0, c.st>>>(
        c.qkvbuf, c.Qb, c.Kb, c.Vb, Nvalid, 0.125f);
    // scores = Q * K^T  (batched over B*H)
    launch_gemm<bf16, true, 0, false>(c.Qb, c.Kb, nullptr, nullptr, c.scores,
        NPAD, NPAD, DHEAD, DHEAD, DHEAD, NPAD,
        BB * NHEAD, (long)NPAD * DHEAD, (long)NPAD * DHEAD, (long)NPAD * NPAD, c.st);
    k_softmax_bf16<<<BB * NHEAD * NPAD, 256, 0, c.st>>>(c.scores, c.attb, Nvalid);
    // out = att * V
    launch_gemm<bf16, false, 0, false>(c.attb, c.Vb, nullptr, nullptr, c.obuf,
        NPAD, DHEAD, NPAD, NPAD, DHEAD, DHEAD,
        BB * NHEAD, (long)NPAD * NPAD, (long)NPAD * DHEAD, (long)NPAD * DHEAD, c.st);
    k_merge_heads<<<(unsigned)((mhTot + 255) / 256), 256, 0, c.st>>>(c.obuf, c.ctxbuf, Nvalid);
    // x = ctx @ proj_w + proj_b + x   (in-place residual: one writer per element)
    launch_gemm<float, false, 0, true>(c.ctxbuf, c.wproj + (long)l * DD * DD,
        c.proj_b + l * DD, c.x, c.x, M, DD, DD, DD, DD, DD, 1, 0, 0, 0, c.st);
    k_layernorm<<<M, 256, 0, c.st>>>(c.x, c.ln2_g + l * DD, c.ln2_b + l * DD, c.h, DD);
    launch_gemm<float, false, 1, false>(c.h, c.wfc1 + (long)l * DD * DFFN,
        c.fc1_b + l * DFFN, nullptr, c.mid, M, DFFN, DD, DD, DFFN, DFFN, 1, 0, 0, 0, c.st);
    launch_gemm<float, false, 0, true>(c.mid, c.wfc2 + (long)l * DFFN * DD,
        c.fc2_b + l * DD, c.x, c.x, M, DD, DFFN, DFFN, DD, DD, 1, 0, 0, 0, c.st);
  }
}

// ---------------------------------------------------------------------------
// kernel_launch
// ---------------------------------------------------------------------------
extern "C" void kernel_launch(void* const* d_in, const int* in_sizes, int n_in,
                              void* d_out, int out_size, void* d_ws, size_t ws_size,
                              hipStream_t stream) {
  (void)in_sizes; (void)n_in; (void)out_size; (void)ws_size;
  const float* in_inputs  = (const float*)d_in[0];
  const float* patch_w    = (const float*)d_in[1];
  const float* patch_b    = (const float*)d_in[2];
  const float* cls_token  = (const float*)d_in[3];
  const float* pos_embed  = (const float*)d_in[4];
  const float* ln1_g      = (const float*)d_in[5];
  const float* ln1_b      = (const float*)d_in[6];
  const float* qkv_w      = (const float*)d_in[7];
  const float* qkv_b      = (const float*)d_in[8];
  const float* proj_w     = (const float*)d_in[9];
  const float* proj_b     = (const float*)d_in[10];
  const float* ln2_g      = (const float*)d_in[11];
  const float* ln2_b      = (const float*)d_in[12];
  const float* fc1_w      = (const float*)d_in[13];
  const float* fc1_b      = (const float*)d_in[14];
  const float* fc2_w      = (const float*)d_in[15];
  const float* fc2_b      = (const float*)d_in[16];
  const float* norm_g     = (const float*)d_in[17];
  const float* norm_b     = (const float*)d_in[18];
  const float* head_w     = (const float*)d_in[19];
  const float* head_b     = (const float*)d_in[20];
  const float* prompt     = (const float*)d_in[21];
  const float* meanp      = (const float*)d_in[22];
  const float* variance   = (const float*)d_in[23];

  // ---- workspace layout (bump allocator) ----
  char* w = (char*)d_ws;
  bf16* wqkv   = (bf16*)bump(w, (size_t)NLAY * DD * 3 * DD * sizeof(bf16));
  bf16* wproj  = (bf16*)bump(w, (size_t)NLAY * DD * DD * sizeof(bf16));
  bf16* wfc1   = (bf16*)bump(w, (size_t)NLAY * DD * DFFN * sizeof(bf16));
  bf16* wfc2   = (bf16*)bump(w, (size_t)NLAY * DFFN * DD * sizeof(bf16));
  bf16* wpatchT= (bf16*)bump(w, (size_t)DD * DD * sizeof(bf16));
  bf16* whead  = (bf16*)bump(w, (size_t)DD * NCLS * sizeof(bf16));
  float* Ap    = (float*)bump(w, (size_t)BB * 196 * DD * sizeof(float));
  float* imgTok= (float*)bump(w, (size_t)BB * 196 * DD * sizeof(float));
  float* x     = (float*)bump(w, (size_t)BB * NPAD * DD * sizeof(float));
  float* h     = (float*)bump(w, (size_t)BB * NPAD * DD * sizeof(float));
  float* qkvbuf= (float*)bump(w, (size_t)BB * NPAD * 3 * DD * sizeof(float));
  bf16* Qb     = (bf16*)bump(w, (size_t)BB * NHEAD * NPAD * DHEAD * sizeof(bf16));
  bf16* Kb     = (bf16*)bump(w, (size_t)BB * NHEAD * NPAD * DHEAD * sizeof(bf16));
  bf16* Vb     = (bf16*)bump(w, (size_t)BB * NHEAD * NPAD * DHEAD * sizeof(bf16));
  float* scores= (float*)bump(w, (size_t)BB * NHEAD * NPAD * NPAD * sizeof(float));
  bf16* attb   = (bf16*)bump(w, (size_t)BB * NHEAD * NPAD * NPAD * sizeof(bf16));
  float* obuf  = (float*)bump(w, (size_t)BB * NHEAD * NPAD * DHEAD * sizeof(float));
  float* ctxbuf= (float*)bump(w, (size_t)BB * NPAD * DD * sizeof(float));
  float* mid   = (float*)bump(w, (size_t)BB * NPAD * DFFN * sizeof(float));
  float* Lchol = (float*)bump(w, (size_t)NPOOL * DD * DD * sizeof(float));
  float* logdet= (float*)bump(w, (size_t)NPOOL * sizeof(float));
  float* logp  = (float*)bump(w, (size_t)BB * NPOOL * sizeof(float));
  float* prompts=(float*)bump(w, (size_t)BB * STOP * LPR * DD * sizeof(float));
  float* pooled= (float*)bump(w, (size_t)BB * DD * sizeof(float));

  auto g1 = [](long n) { return (unsigned)((n + 255) / 256); };

  // ---- 1) weights -> bf16 once (170 MB, ~fits 192 MB L2, reused by both passes)
  k_cvt<<<g1((long)NLAY*DD*3*DD), 256, 0, stream>>>(qkv_w, wqkv, (long)NLAY*DD*3*DD);
  k_cvt<<<g1((long)NLAY*DD*DD),   256, 0, stream>>>(proj_w, wproj, (long)NLAY*DD*DD);
  k_cvt<<<g1((long)NLAY*DD*DFFN), 256, 0, stream>>>(fc1_w, wfc1, (long)NLAY*DD*DFFN);
  k_cvt<<<g1((long)NLAY*DFFN*DD), 256, 0, stream>>>(fc2_w, wfc2, (long)NLAY*DFFN*DD);
  k_cvt<<<g1((long)DD*NCLS),      256, 0, stream>>>(head_w, whead, (long)DD*NCLS);
  k_cvt_T<<<g1((long)DD*DD), 256, 0, stream>>>(patch_w, wpatchT, DD, DD);

  // ---- 2) Cholesky of cov = |variance| + I (independent of passes)
  k_cholesky<<<NPOOL, 256, 0, stream>>>(variance, Lchol, logdet);

  // ---- 3) patch embed as GEMM: im2col [B*196,768] x [768,768] + bias
  k_im2col<<<g1((long)BB*196*DD), 256, 0, stream>>>(in_inputs, Ap);
  launch_gemm<float, false, 0, false>(Ap, wpatchT, patch_b, nullptr, imgTok,
      BB * 196, DD, DD, DD, DD, DD, 1, 0, 0, 0, stream);

  // ---- 4) query pass (N=197)
  k_build_x0<<<g1((long)BB*NPAD*DD), 256, 0, stream>>>(imgTok, cls_token, pos_embed, x);
  PassCtx pc = { ln1_g, ln1_b, qkv_b, proj_b, ln2_g, ln2_b, fc1_b, fc2_b,
                 wqkv, wproj, wfc1, wfc2,
                 x, h, qkvbuf, ctxbuf, mid, obuf, scores, Qb, Kb, Vb, attb, stream };
  run_pass(pc, 197);
  k_layernorm<<<BB * NPAD, 256, 0, stream>>>(x, norm_g, norm_b, h, DD);

  // ---- 5) MVN log-prob on cls token + top-k prompt selection
  k_mvn<<<BB * NPOOL, 256, 0, stream>>>(h, (long)NPAD * DD, meanp, Lchol, logdet, logp);
  k_topk_gather<<<BB, 256, 0, stream>>>(logp, prompt, pos_embed, prompts);

  // ---- 6) prompted pass (N=222)
  k_build_x2<<<g1((long)BB*NPAD*DD), 256, 0, stream>>>(imgTok, cls_token, pos_embed, prompts, x);
  run_pass(pc, 222);
  k_layernorm<<<BB * NPAD, 256, 0, stream>>>(x, norm_g, norm_b, h, DD);

  // ---- 7) prompt-token mean pool + head GEMM -> d_out [16,100] f32
  k_pool<<<g1((long)BB*DD), 256, 0, stream>>>(h, pooled);
  launch_gemm<float, false, 0, false>(pooled, whead, head_b, nullptr, (float*)d_out,
      BB, NCLS, DD, DD, NCLS, NCLS, 1, 0, 0, 0, stream);
}
